// RecursiveRotorAccumulator_12146167513839
// MI455X (gfx1250) — matrix operand order repaired and checked by reading, not compiled, not run on hardware
//
#include <hip/hip_runtime.h>

// Problem constants from the reference
#define S_LEN   512
#define D_DIM   256
#define MV      32
#define ECHUNK  16          // channels (e) per block
#define XPAD    264         // padded d-stride for LDS (multiple of 8, dodges bank conflicts)

typedef __attribute__((ext_vector_type(16))) __bf16 bf16x16;
typedef __attribute__((ext_vector_type(8)))  float  f32x8;

// ---------------------------------------------------------------------------
// Cayley sign table for Cl(4,1): e_a * e_b = sign(a,b) * e_{a^b}
// NEG.m[a] bit k == 1  <=>  sign(a, a^k) == -1
// ---------------------------------------------------------------------------
struct NegTab { unsigned m[32]; };

__host__ __device__ constexpr int gp_sign(int a, int bb) {
    int sign = 1;
    int t = a >> 1;
    while (t) {                       // reorder (swap-count) parity
        int p = t & bb, c = 0;
        while (p) { c ^= (p & 1); p >>= 1; }
        if (c) sign = -sign;
        t >>= 1;
    }
    if ((a & bb) & 0x10) sign = -sign;  // metric (1,1,1,1,-1): only e5^2 = -1
    return sign;
}

__host__ __device__ constexpr NegTab make_neg() {
    NegTab t{};
    for (int a = 0; a < 32; ++a) {
        unsigned mm = 0u;
        for (int k = 0; k < 32; ++k)
            if (gp_sign(a, a ^ k) < 0) mm |= (1u << k);
        t.m[a] = mm;
    }
    return t;
}

__device__ constexpr NegTab NEG = make_neg();

// ---------------------------------------------------------------------------
// Fused VersorLinear + recursive rotor scan.
// Block = (batch b, 16-channel chunk e0). 512 threads = 16 waves.
// Wave c <-> channel e0+c ; lane k <-> blade k. psi lives in one VGPR/lane.
// Per step t:
//   1) all waves stage x[b,t,:,:] (256x32 f32) -> bf16 LDS (transposed [k][d])
//   2) waves 0..1 compute delta[16x32] = Wchunk(16x256) @ X(256x32) + bias
//      via v_wmma_f32_16x16x32_bf16; A-fragments (W) are hoisted to registers
//      once since W is step-invariant — only B-fragments come from LDS.
//   3) every wave: geometric product via xor-shuffles + sign masks, then
//      L2-normalize over the 32 lanes (butterfly), all in registers.
// ---------------------------------------------------------------------------
__global__ __launch_bounds__(512)
void rotor_scan_kernel(const float* __restrict__ x,
                       const float* __restrict__ W,
                       const float* __restrict__ bias,
                       float* __restrict__ out)
{
    __shared__ __bf16 Xs[MV][XPAD];        // Xs[k][d]  (B-matrix source)
    __shared__ __bf16 Ws[ECHUNK][XPAD];    // Ws[m][d]  (A-matrix staging)
    __shared__ float  dbuf[ECHUNK][MV + 1];
    __shared__ float  bias_s[ECHUNK];

    const int tid  = threadIdx.x;
    const int wv   = tid >> 5;             // wave id 0..15
    const int lane = tid & 31;             // lane within wave32
    const int b    = blockIdx.y;
    const int e0   = blockIdx.x * ECHUNK;

    const int half16 = lane >> 4;          // 0: lanes 0-15, 1: lanes 16-31
    const int l15    = lane & 15;

    // Stage W chunk as bf16 (once) and bias
    for (int idx = tid; idx < ECHUNK * D_DIM; idx += blockDim.x) {
        int m = idx >> 8, d = idx & (D_DIM - 1);
        Ws[m][d] = (__bf16)W[(e0 + m) * D_DIM + d];
    }
    if (tid < ECHUNK) bias_s[tid] = bias[e0 + tid];
    __syncthreads();                       // Ws ready

    // Hoist step-invariant A fragments (W chunk) into registers.
    // A 16x32 bf16 layout (ISA 7.12.2): row M = l15; lanes<16 hold K 0-7 & 16-23,
    // lanes>=16 hold K 8-15 & 24-31.
    bf16x16 afrag[8];
    if (wv < 2) {
        #pragma unroll
        for (int kk = 0; kk < 8; ++kk) {
            #pragma unroll
            for (int i = 0; i < 8; ++i) {
                afrag[kk][i]     = Ws[l15][kk * 32 + half16 * 8 + i];
                afrag[kk][i + 8] = Ws[l15][kk * 32 + 16 + half16 * 8 + i];
            }
        }
    }

    float psi = (lane == 0) ? 1.0f : 0.0f; // psi0 = scalar blade e0

    const float* xb = x + (size_t)b * S_LEN * D_DIM * MV;

    for (int t = 0; t < S_LEN; ++t) {
        __syncthreads();                   // step t-1 consumers of Xs/dbuf done

        // ---- stage x[b,t] -> Xs (bf16, transposed) ----
        const float* xt = xb + (size_t)t * D_DIM * MV;
        #pragma unroll
        for (int j = 0; j < 16; ++j) {
            int d = wv * 16 + j;           // wave reads 32 consecutive k: coalesced 128B
            Xs[lane][d] = (__bf16)xt[d * MV + lane];
        }
        if (t + 1 < S_LEN) {               // CDNA5 global_prefetch_b8 for next tile
            __builtin_prefetch(xt + D_DIM * MV + (wv * 16) * MV + lane, 0, 0);
        }
        __syncthreads();                   // Xs ready

        // ---- delta tile: waves 0,1 each own one 16-wide blade tile ----
        if (wv < 2) {
            const int kcol = l15 + wv * 16;     // global blade column (B-matrix N=l15)
            f32x8 acc = {};
            #pragma unroll
            for (int kk = 0; kk < 8; ++kk) {
                // B 32x16 bf16 layout: col N = l15; lanes<16 K 0-7 & 16-23, lanes>=16 K 8-15 & 24-31
                bf16x16 bm;
                #pragma unroll
                for (int i = 0; i < 16; ++i)
                    bm[i] = Xs[kcol][kk * 32 + (i >> 3) * 16 + half16 * 8 + (i & 7)];
                acc = __builtin_amdgcn_wmma_f32_16x16x32_bf16(
                        false, afrag[kk], false, bm, (short)0, acc, false, false);
            }
            // C/D layout: VGPR r -> M = r + 8*half16, N = l15
            #pragma unroll
            for (int r = 0; r < 8; ++r) {
                int m = r + half16 * 8;
                dbuf[m][kcol] = acc[r] + bias_s[m];
            }
        }
        __syncthreads();                   // dbuf ready

        // ---- geometric product + normalize (registers + wave shuffles) ----
        float rr = dbuf[wv][lane];         // r_t[blade=lane] for channel e0+wv

        float nk = 0.0f;
        #pragma unroll
        for (int a = 0; a < 32; ++a) {
            float ra = __shfl(rr, a, 32);          // broadcast r[a]
            float pb = __shfl_xor(psi, a, 32);     // lane k gets psi[k^a]
            unsigned nbit = (NEG.m[a] >> lane) & 1u;
            int pv = __float_as_int(pb) ^ (int)(nbit << 31);   // apply Cayley sign
            nk = fmaf(ra, __int_as_float(pv), nk);
        }
        float s = nk * nk;                 // butterfly L2 reduction over blades
        #pragma unroll
        for (int off = 16; off > 0; off >>= 1) s += __shfl_xor(s, off, 32);
        psi = nk * rsqrtf(s + 1e-12f);
    }

    // out shape (B, D, 32)
    out[((size_t)b * D_DIM + (e0 + wv)) * MV + lane] = psi;
}

extern "C" void kernel_launch(void* const* d_in, const int* in_sizes, int n_in,
                              void* d_out, int out_size, void* d_ws, size_t ws_size,
                              hipStream_t stream) {
    (void)n_in; (void)out_size; (void)d_ws; (void)ws_size;
    const float* x = (const float*)d_in[0];   // (B,S,D,32) f32
    const float* W = (const float*)d_in[1];   // (D,D)      f32
    const float* b = (const float*)d_in[2];   // (D,)       f32
    float* out = (float*)d_out;               // (B,D,32)   f32

    const int Bn = in_sizes[0] / (S_LEN * D_DIM * MV);
    dim3 grid(D_DIM / ECHUNK, Bn);
    rotor_scan_kernel<<<grid, 512, 0, stream>>>(x, W, b, out);
}
